// Encoder_9251359556329
// MI455X (gfx1250) — matrix operand order repaired and checked by reading, not compiled, not run on hardware
//
#include <hip/hip_runtime.h>
#include <hip/hip_bf16.h>

// ---------------------------------------------------------------------------
// Types for CDNA5 WMMA bf16: A/B = 16 bf16 per lane (8 VGPRs), C/D = 8 f32.
// ---------------------------------------------------------------------------
typedef __bf16 bf16_t;
typedef bf16_t v16bf __attribute__((ext_vector_type(16)));
typedef float  v8f   __attribute__((ext_vector_type(8)));

union AB16 { v16bf v; unsigned u[8]; };

// A-matrix 16x32 bf16 fragment (row-major source, row stride in elements).
// ISA layout: lanes 0-15 & 16-31 both hold M=0..15; VGPR v<4: K = grp*8 + 2v,
// VGPR v>=4: K = 16 + grp*8 + 2(v-4).  K-pairs are contiguous -> dword loads.
__device__ __forceinline__ v16bf load_a_frag(const bf16_t* base, int row_stride, int lane) {
    int ln = lane & 15, grp = lane >> 4;
    const bf16_t* p = base + (size_t)ln * row_stride;
    AB16 r;
#pragma unroll
    for (int i = 0; i < 4; ++i)
        r.u[i] = *(const unsigned*)(p + grp * 8 + 2 * i);
#pragma unroll
    for (int i = 0; i < 4; ++i)
        r.u[4 + i] = *(const unsigned*)(p + 16 + grp * 8 + 2 * i);
    return r.v;
}

// B-matrix 32x16 bf16 fragment from a PRE-TRANSPOSED weight (N x K row-major).
// ISA layout: lane n = ln; VGPR v<4: K = grp*8 + 2v, VGPR v>=4:
// K = 16 + grp*8 + 2(v-4) (+kb).  Contiguous K-pairs -> dword loads.
__device__ __forceinline__ v16bf load_b_frag(const bf16_t* base, int row_stride, int lane) {
    int ln = lane & 15, grp = lane >> 4;
    const bf16_t* p = base + (size_t)ln * row_stride + grp * 8;
    AB16 r;
#pragma unroll
    for (int i = 0; i < 8; ++i)
        r.u[i] = *(const unsigned*)(p + (i >> 2) * 16 + 2 * (i & 3));
    return r.v;
}

__device__ __forceinline__ v8f wmma_bf16(v16bf a, v16bf b, v8f c) {
    return __builtin_amdgcn_wmma_f32_16x16x32_bf16(false, a, false, b, (short)0, c, false, false);
}

__device__ __forceinline__ float sigmoidf_(float x) { return 1.0f / (1.0f + __expf(-x)); }

// gfx1250 async global -> LDS copy (16B per lane), tracked by ASYNCcnt.
// lds_off = byte offset inside the workgroup LDS allocation (low 32 bits of a
// generic shared-memory address), gaddr = 64-bit global address.
__device__ __forceinline__ void async_copy_b128(unsigned lds_off, const void* gaddr) {
    asm volatile("global_load_async_to_lds_b128 %0, %1, off"
                 :: "v"(lds_off), "v"((unsigned long long)(uintptr_t)gaddr)
                 : "memory");
}
__device__ __forceinline__ void wait_asynccnt0() {
    asm volatile("s_wait_asynccnt 0" ::: "memory");
}

// Grid-wide split barrier: monotone counter, one arrival per block per step.
__device__ __forceinline__ void grid_sync(unsigned* bar, unsigned expected) {
    __syncthreads();
    if (threadIdx.x == 0) {
        __threadfence();
        atomicAdd(bar, 1u);
        while (*(volatile unsigned*)bar < expected) { __builtin_amdgcn_s_sleep(1); }
        __threadfence();
    }
    __syncthreads();
}

// ---------------------------------------------------------------------------
// Embedding gather + fp32 -> bf16
// ---------------------------------------------------------------------------
__global__ __launch_bounds__(256) void k_embed(const int* __restrict__ tokens,
                                               const float* __restrict__ emb,
                                               bf16_t* __restrict__ xb, int E) {
    int row = blockIdx.x;
    int tok = tokens[row];
    const float* e = emb + (size_t)tok * E;
    bf16_t* o = xb + (size_t)row * E;
    for (int i = threadIdx.x; i < E; i += blockDim.x)
        o[i] = (bf16_t)e[i];
}

// Weight convert + transpose: W (K x N, f32) -> Wt (N x K, bf16)
__global__ __launch_bounds__(256) void k_cvtw(const float* __restrict__ w,
                                              bf16_t* __restrict__ wt, int K, int N) {
    size_t idx = (size_t)blockIdx.x * blockDim.x + threadIdx.x;
    if (idx >= (size_t)K * N) return;
    int k = (int)(idx / N), n = (int)(idx % N);
    wt[(size_t)n * K + k] = (bf16_t)w[idx];
}

// ---------------------------------------------------------------------------
// C(M,N) = A(M,K bf16) @ Bt(N,K bf16)^T + bias(N), fp32 out.
// 8 waves/block; each wave owns one 16x16 tile; block covers 32(M) x 64(N).
// ---------------------------------------------------------------------------
__global__ __launch_bounds__(256) void k_gemm_bias(const bf16_t* __restrict__ A,
                                                   const bf16_t* __restrict__ Bt,
                                                   const float* __restrict__ bias,
                                                   float* __restrict__ C,
                                                   int M, int N, int K) {
    int lane = threadIdx.x & 31;
    int wave = threadIdx.x >> 5;
    int mt = wave >> 2, nt = wave & 3;
    int m0 = (blockIdx.y * 2 + mt) * 16;
    int n0 = (blockIdx.x * 4 + nt) * 16;
    int ln = lane & 15, grp = lane >> 4;

    v8f acc;
    float bv = bias[n0 + ln];
#pragma unroll
    for (int r = 0; r < 8; ++r) acc[r] = bv;

    const bf16_t* ap = A + (size_t)m0 * K;
    const bf16_t* bp = Bt + (size_t)n0 * K;
    for (int kb = 0; kb < K; kb += 32) {
        __builtin_prefetch(ap + kb + 64, 0, 1);
        __builtin_prefetch(bp + kb + 64, 0, 1);
        v16bf a = load_a_frag(ap + kb, K, lane);
        v16bf b = load_b_frag(bp + kb, K, lane);
        acc = wmma_bf16(a, b, acc);
    }

    float* cp = C + (size_t)(m0 + grp * 8) * N + n0 + ln;
#pragma unroll
    for (int r = 0; r < 8; ++r) cp[(size_t)r * N] = acc[r];
}

// ---------------------------------------------------------------------------
// Persistent bidirectional LSTM scan. 32 blocks x 256 threads.
// Blocks 0-15: forward dir, 16-31: backward dir. 128 waves/dir, one 16x16
// h-tile per wave (M=64 -> 4 tiles, U=512 -> 32 tiles). All 8 waves of a
// block share the same M-tile, so the 16x512 bf16 h_prev tile (16KB) is
// staged once per step into LDS with async global->LDS copies; A fragments
// are then read via DS. 4 gate WMMAs share one A fragment per K chunk.
// Grid barrier between time steps.
// layer==0: h kept time-indexed in out0 bf16 (B,T,2U). layer==1: h in
// h1f/h1b bf16 buffers; fp32 h -> xout; final h,c -> eh/ec.
// ---------------------------------------------------------------------------
__global__ __launch_bounds__(256) void k_lstm(const float* __restrict__ xz_f,
                                              const float* __restrict__ xz_b,
                                              const bf16_t* __restrict__ Whf,
                                              const bf16_t* __restrict__ Whb,
                                              bf16_t* __restrict__ out0,
                                              bf16_t* __restrict__ h1f,
                                              bf16_t* __restrict__ h1b,
                                              float* __restrict__ xout,
                                              float* __restrict__ eh,
                                              float* __restrict__ ec,
                                              unsigned* bar, int layer, int T) {
    const int U = 512, G = 2048;
    __shared__ bf16_t sh[16 * 512];          // 16KB staged h_prev tile

    int lane = threadIdx.x & 31;
    int wave = threadIdx.x >> 5;
    int dir  = blockIdx.x >> 4;
    int blk  = blockIdx.x & 15;
    int wid  = blk * 8 + wave;            // 0..127
    int mt = wid >> 5, nt = wid & 31;     // all 8 waves of a block share mt
    int m0 = mt * 16, n0 = nt * 16;
    int ln = lane & 15, grp = lane >> 4;

    const float*  xz = dir ? xz_b : xz_f;
    const bf16_t* Wh = dir ? Whb : Whf;
    bf16_t* hbuf = dir ? h1b : h1f;

    float c[8];
#pragma unroll
    for (int r = 0; r < 8; ++r) c[r] = 0.0f;

    unsigned nb = gridDim.x;
    unsigned lds0 = (unsigned)(uintptr_t)(&sh[0]);

    for (int s = 0; s < T; ++s) {
        int t = dir ? (T - 1 - s) : s;

        // Gate accumulators start from precomputed x@Wx + b.
        v8f az[4];
#pragma unroll
        for (int g = 0; g < 4; ++g) {
            const float* zp = xz + ((size_t)(m0 + grp * 8) * T + t) * G + g * U + n0 + ln;
#pragma unroll
            for (int r = 0; r < 8; ++r) az[g][r] = zp[(size_t)r * T * G];
        }

        if (s > 0) {
            // ---- stage shared h_prev tile (16 rows x 512 cols bf16) to LDS ----
            int tp = dir ? (t + 1) : (t - 1);
            const bf16_t* rowbase;
            size_t row_stride;
            if (layer == 0) {
                rowbase = out0 + ((size_t)m0 * T + tp) * (2 * U) + dir * U;
                row_stride = (size_t)T * 2 * U;
            } else {
                rowbase = hbuf + (size_t)m0 * U;
                row_stride = U;
            }
            // 16 rows x 1KB = 1024 x 16B chunks; 256 threads x 4 chunks.
#pragma unroll
            for (int it = 0; it < 4; ++it) {
                int chunk = it * 256 + threadIdx.x;
                int row = chunk >> 6;            // 64 x 16B chunks per row
                int col = chunk & 63;
                const void* g = rowbase + (size_t)row * row_stride + col * 8;
                async_copy_b128(lds0 + (unsigned)chunk * 16, g);
            }
            wait_asynccnt0();
            __syncthreads();

            // ---- K loop: A from LDS (shared), B (Wh) streamed from L2 ----
            for (int kb = 0; kb < U; kb += 32) {
                v16bf a = load_a_frag(sh + kb, U, lane);
#pragma unroll
                for (int g = 0; g < 4; ++g) {
                    v16bf b = load_b_frag(Wh + (size_t)(g * U + n0) * U + kb, U, lane);
                    az[g] = wmma_bf16(a, b, az[g]);
                }
            }
        }

        // Elementwise LSTM cell update on the 16x16 tile (8 values per lane).
#pragma unroll
        for (int r = 0; r < 8; ++r) {
            float iv = sigmoidf_(az[0][r]);
            float fv = sigmoidf_(az[1][r]);
            float gv = tanhf(az[2][r]);
            float ov = sigmoidf_(az[3][r]);
            c[r] = fv * c[r] + iv * gv;
            float hv = ov * tanhf(c[r]);
            int m = m0 + grp * 8 + r;
            int n = n0 + ln;
            if (layer == 0) {
                out0[((size_t)m * T + t) * (2 * U) + dir * U + n] = (bf16_t)hv;
            } else {
                hbuf[(size_t)m * U + n] = (bf16_t)hv;
                xout[((size_t)m * T + t) * (2 * U) + dir * U + n] = hv;
                if (s == T - 1) {
                    eh[(size_t)m * (2 * U) + dir * U + n] = hv;
                    ec[(size_t)m * (2 * U) + dir * U + n] = c[r];
                }
            }
        }

        grid_sync(bar, nb * (unsigned)(s + 1));
    }
}

// ---------------------------------------------------------------------------
// Host side
// ---------------------------------------------------------------------------
extern "C" void kernel_launch(void* const* d_in, const int* in_sizes, int n_in,
                              void* d_out, int out_size, void* d_ws, size_t ws_size,
                              hipStream_t stream) {
    const int B = 64, T = 256, E = 512, U = 512, G = 2048;
    const int M = B * T;

    const int*   tokens = (const int*)d_in[0];
    const float* emb    = (const float*)d_in[1];
    const float* Wx_f0 = (const float*)d_in[2];
    const float* Wh_f0 = (const float*)d_in[3];
    const float* b_f0  = (const float*)d_in[4];
    const float* Wx_b0 = (const float*)d_in[5];
    const float* Wh_b0 = (const float*)d_in[6];
    const float* b_b0  = (const float*)d_in[7];
    const float* Wx_f1 = (const float*)d_in[8];
    const float* Wh_f1 = (const float*)d_in[9];
    const float* b_f1  = (const float*)d_in[10];
    const float* Wx_b1 = (const float*)d_in[11];
    const float* Wh_b1 = (const float*)d_in[12];
    const float* b_b1  = (const float*)d_in[13];

    float* out_x = (float*)d_out;                       // (B,T,2U)
    float* out_h = out_x + (size_t)M * 2 * U;           // (B,2U)
    float* out_c = out_h + (size_t)B * 2 * U;           // (B,2U)

    char* ws = (char*)d_ws;
    size_t off = 0;
    auto alloc = [&](size_t bytes) -> void* {
        void* p = ws + off;
        off += (bytes + 255) & ~(size_t)255;
        return p;
    };
    unsigned* bar  = (unsigned*)alloc(256);                     // 2 counters used
    bf16_t* xbf    = (bf16_t*)alloc((size_t)M * E * 2);         // 16 MB
    bf16_t* wxf0t  = (bf16_t*)alloc((size_t)E * G * 2);
    bf16_t* whf0t  = (bf16_t*)alloc((size_t)U * G * 2);
    bf16_t* wxb0t  = (bf16_t*)alloc((size_t)E * G * 2);
    bf16_t* whb0t  = (bf16_t*)alloc((size_t)U * G * 2);
    bf16_t* wxf1t  = (bf16_t*)alloc((size_t)2 * U * G * 2);
    bf16_t* whf1t  = (bf16_t*)alloc((size_t)U * G * 2);
    bf16_t* wxb1t  = (bf16_t*)alloc((size_t)2 * U * G * 2);
    bf16_t* whb1t  = (bf16_t*)alloc((size_t)U * G * 2);
    float* xzf     = (float*)alloc((size_t)M * G * 4);          // 128 MB
    float* xzb     = (float*)alloc((size_t)M * G * 4);          // 128 MB
    bf16_t* out0   = (bf16_t*)alloc((size_t)M * 2 * U * 2);     // 32 MB
    bf16_t* h1f    = (bf16_t*)alloc((size_t)B * U * 2);
    bf16_t* h1b    = (bf16_t*)alloc((size_t)B * U * 2);

    hipMemsetAsync(bar, 0, 256, stream);

    k_embed<<<M, 256, 0, stream>>>(tokens, emb, xbf, E);

    auto cvt = [&](const float* w, bf16_t* wt, int K, int N) {
        size_t n = (size_t)K * N;
        k_cvtw<<<(unsigned)((n + 255) / 256), 256, 0, stream>>>(w, wt, K, N);
    };
    cvt(Wx_f0, wxf0t, E, G);     cvt(Wh_f0, whf0t, U, G);
    cvt(Wx_b0, wxb0t, E, G);     cvt(Wh_b0, whb0t, U, G);
    cvt(Wx_f1, wxf1t, 2 * U, G); cvt(Wh_f1, whf1t, U, G);
    cvt(Wx_b1, wxb1t, 2 * U, G); cvt(Wh_b1, whb1t, U, G);

    dim3 gg(G / 64, M / 32);
    // Layer 0 input projections
    k_gemm_bias<<<gg, 256, 0, stream>>>(xbf, wxf0t, b_f0, xzf, M, G, E);
    k_gemm_bias<<<gg, 256, 0, stream>>>(xbf, wxb0t, b_b0, xzb, M, G, E);
    // Layer 0 bidirectional scan -> out0 (bf16, (B,T,2U))
    k_lstm<<<32, 256, 0, stream>>>(xzf, xzb, whf0t, whb0t, out0, h1f, h1b,
                                   out_x, out_h, out_c, bar, 0, T);
    // Layer 1 input projections (reuse xz buffers)
    k_gemm_bias<<<gg, 256, 0, stream>>>(out0, wxf1t, b_f1, xzf, M, G, 2 * U);
    k_gemm_bias<<<gg, 256, 0, stream>>>(out0, wxb1t, b_b1, xzb, M, G, 2 * U);
    // Layer 1 bidirectional scan -> fp32 outputs + final states
    k_lstm<<<32, 256, 0, stream>>>(xzf, xzb, whf1t, whb1t, out0, h1f, h1b,
                                   out_x, out_h, out_c, bar + 64, 1, T);
}
